// WindowAttentiondwconve2_22187801051586
// MI455X (gfx1250) — matrix operand order repaired and checked by reading, not compiled, not run on hardware
//
#include <hip/hip_runtime.h>

// ---- POD vector types (avoid HIP's ctor-ful uint4 inside unions) ----
typedef unsigned int u32;
typedef __attribute__((ext_vector_type(4))) u32   u32x4;
typedef __attribute__((ext_vector_type(2))) u32   u32x2;
typedef __attribute__((ext_vector_type(4))) float f32x4;
typedef __attribute__((ext_vector_type(16))) __bf16 v16bf;
typedef __attribute__((ext_vector_type(8)))  float  v8f;

union U16BF { v16bf v; u32x4 u[2]; };
union PK4   { __bf16 b[4]; u32x2 u; };

// =====================================================================
// GEMM: C[M,512] = A[M,512] (f32 or bf16) x W[512,512-slice] (f32) + bias
// Block tile 128x128, BK=32, 256 threads = 8 waves (4 along M, 2 along N),
// each wave computes 32x64 via 2x4 v_wmma_f32_16x16x32_bf16 accumulators.
// Double-buffered LDS: prefetch tile k+1 into regs while WMMAs run tile k.
// =====================================================================
template <typename AT, typename OT>
__global__ __launch_bounds__(256) void gemm_k512(
    const AT* __restrict__ A,            // [M,512]
    const float* __restrict__ W,         // [512, ldw] (column slice base)
    int ldw,
    const float* __restrict__ bias,      // [512] slice base
    OT* __restrict__ C)                  // [M,512]
{
    constexpr int LDA = 512, LDC = 512, BK = 32, NSTEP = 16, STR = 40; // 80B row stride, 16B aligned
    __shared__ __bf16 As[2][128 * STR];  // [row][k]
    __shared__ __bf16 Bs[2][128 * STR];  // [n][k]  (W transposed)

    const int tid  = threadIdx.x;
    const int lane = tid & 31;
    const int wid  = tid >> 5;
    const int lg   = lane >> 4;          // half-wave group
    const int lm   = lane & 15;
    const int wm   = wid & 3;            // 4 waves along M
    const int wn   = wid >> 2;           // 2 waves along N
    const int m0   = blockIdx.y * 128;
    const int n0   = blockIdx.x * 128;

    v8f acc[2][4];
#pragma unroll
    for (int i = 0; i < 2; ++i)
#pragma unroll
        for (int j = 0; j < 4; ++j) acc[i][j] = {};

    u32x2 ra[4];   // packed bf16 x4 A-tile prefetch
    f32x4 rb[4];   // f32 W-tile prefetch

    auto loadTile = [&](int k0) {
#pragma unroll
        for (int i = 0; i < 4; ++i) {
            int f   = tid + i * 256;     // 0..1023, 8 chunks of 4 per A row
            int row = f >> 3;
            int cc  = (f & 7) * 4;
            if constexpr (sizeof(AT) == 4) {
                f32x4 v = *(const f32x4*)(A + (size_t)(m0 + row) * LDA + k0 + cc);
                PK4 p;
                p.b[0] = (__bf16)v[0]; p.b[1] = (__bf16)v[1];
                p.b[2] = (__bf16)v[2]; p.b[3] = (__bf16)v[3];
                ra[i] = p.u;
            } else {
                ra[i] = *(const u32x2*)(A + (size_t)(m0 + row) * LDA + k0 + cc);
            }
        }
#pragma unroll
        for (int i = 0; i < 4; ++i) {
            int f  = tid + i * 256;      // 32 chunks of 4 per W k-row
            int kk = f >> 5;
            int nn = (f & 31) * 4;
            rb[i] = *(const f32x4*)(W + (size_t)(k0 + kk) * ldw + n0 + nn);
        }
    };
    auto stageTile = [&](int buf) {
#pragma unroll
        for (int i = 0; i < 4; ++i) {
            int f   = tid + i * 256;
            int row = f >> 3;
            int cc  = (f & 7) * 4;
            *(u32x2*)&As[buf][row * STR + cc] = ra[i];
        }
#pragma unroll
        for (int i = 0; i < 4; ++i) {
            int f  = tid + i * 256;
            int kk = f >> 5;
            int nn = (f & 31) * 4;
            Bs[buf][(nn + 0) * STR + kk] = (__bf16)rb[i][0];
            Bs[buf][(nn + 1) * STR + kk] = (__bf16)rb[i][1];
            Bs[buf][(nn + 2) * STR + kk] = (__bf16)rb[i][2];
            Bs[buf][(nn + 3) * STR + kk] = (__bf16)rb[i][3];
        }
    };

    loadTile(0);
    stageTile(0);
    __syncthreads();

    for (int s = 0; s < NSTEP; ++s) {
        const int cur = s & 1;
        if (s + 1 < NSTEP) loadTile((s + 1) * BK);   // global -> regs (overlaps WMMA below)

        v16bf af[2], bfr[4];
#pragma unroll
        for (int mt = 0; mt < 2; ++mt) {
            int row = wm * 32 + mt * 16 + lm;
            U16BF t;
            t.u[0] = *(const u32x4*)&As[cur][row * STR + 8 * lg];
            t.u[1] = *(const u32x4*)&As[cur][row * STR + 16 + 8 * lg];
            af[mt] = t.v;
        }
#pragma unroll
        for (int nt = 0; nt < 4; ++nt) {
            int col = wn * 64 + nt * 16 + lm;
            U16BF t;
            t.u[0] = *(const u32x4*)&Bs[cur][col * STR + 8 * lg];
            t.u[1] = *(const u32x4*)&Bs[cur][col * STR + 16 + 8 * lg];
            bfr[nt] = t.v;
        }
#pragma unroll
        for (int mt = 0; mt < 2; ++mt)
#pragma unroll
            for (int nt = 0; nt < 4; ++nt)
                acc[mt][nt] = __builtin_amdgcn_wmma_f32_16x16x32_bf16(
                    false, af[mt], false, bfr[nt], (short)0, acc[mt][nt], false, false);

        if (s + 1 < NSTEP) {
            stageTile(cur ^ 1);                      // regs -> other LDS buffer
            __syncthreads();
        }
    }

    // ---- epilogue: + bias, store ----
#pragma unroll
    for (int mt = 0; mt < 2; ++mt) {
#pragma unroll
        for (int nt = 0; nt < 4; ++nt) {
            int col  = n0 + wn * 64 + nt * 16 + lm;
            float bv = bias[col];
#pragma unroll
            for (int r = 0; r < 8; ++r) {
                int m   = m0 + wm * 32 + mt * 16 + r + 8 * lg;
                float o = acc[mt][nt][r] + bv;
                if constexpr (sizeof(OT) == 2)
                    C[(size_t)m * LDC + col] = (__bf16)o;
                else
                    C[(size_t)m * LDC + col] = o;
            }
        }
    }
}

// =====================================================================
// Attention per (b, h): S = (q*scale*gate) K^T (49x49 padded to 64x64),
// row softmax, O = P V + a*V. 128 threads = 4 waves (one 16-row M tile each).
// V loaded coalesced, transposed through LDS (no scattered global traffic).
// =====================================================================
__global__ __launch_bounds__(128) void attn49(
    const __bf16* __restrict__ qb, const __bf16* __restrict__ gb,
    const __bf16* __restrict__ kb, const __bf16* __restrict__ vb,
    const float* __restrict__ aptr, __bf16* __restrict__ ob)
{
    constexpr int   N     = 49;
    constexpr float SCALE = 0.17677669529663687f;  // 32^-0.5
    __shared__ __bf16 Qs[64 * 40];   // q*scale*gate, [row][d]
    __shared__ __bf16 Ks[64 * 40];   // [key][d]
    __shared__ __bf16 Vr[64 * 40];   // V row-major [key][d] (coalesced load)
    __shared__ float  Sf[64 * 66];   // scores
    __shared__ __bf16 Ps[64 * 72];   // softmax probs [row][key]
    __shared__ __bf16 Vt[32 * 72];   // V transposed [d][key]

    const int tid  = threadIdx.x;
    const int lane = tid & 31;
    const int w    = tid >> 5;
    const int lg   = lane >> 4;
    const int lm   = lane & 15;
    const int b    = blockIdx.x >> 4;
    const int h    = blockIdx.x & 15;
    const size_t tokbase = (size_t)b * N;
    const int ch   = h * 32;

    // ---- load Q*g*scale, K, V rows (64x32 each, zero padded) ----
    {
        int r  = tid >> 1;
        int hh = (tid & 1) * 16;
        if (r < N) {
            union { __bf16 e[16]; u32x4 u[2]; } q, g, k, v;
            const __bf16* qp = qb + (tokbase + r) * 512 + ch + hh;
            const __bf16* gp = gb + (tokbase + r) * 512 + ch + hh;
            const __bf16* kp = kb + (tokbase + r) * 512 + ch + hh;
            const __bf16* vp = vb + (tokbase + r) * 512 + ch + hh;
            q.u[0] = *(const u32x4*)qp;  q.u[1] = *(const u32x4*)(qp + 8);
            g.u[0] = *(const u32x4*)gp;  g.u[1] = *(const u32x4*)(gp + 8);
            k.u[0] = *(const u32x4*)kp;  k.u[1] = *(const u32x4*)(kp + 8);
            v.u[0] = *(const u32x4*)vp;  v.u[1] = *(const u32x4*)(vp + 8);
#pragma unroll
            for (int i = 0; i < 16; ++i) {
                Qs[r * 40 + hh + i] = (__bf16)((float)q.e[i] * (float)g.e[i] * SCALE);
                Ks[r * 40 + hh + i] = k.e[i];
                Vr[r * 40 + hh + i] = v.e[i];
            }
        } else {
#pragma unroll
            for (int i = 0; i < 16; ++i) {
                Qs[r * 40 + hh + i] = (__bf16)0.0f;
                Ks[r * 40 + hh + i] = (__bf16)0.0f;
                Vr[r * 40 + hh + i] = (__bf16)0.0f;
            }
        }
    }
    __syncthreads();

    // ---- S = Q K^T : each wave does one 16-row tile x 4 key tiles ----
    {
        int mrow = w * 16 + lm;
        U16BF ta;
        ta.u[0] = *(const u32x4*)&Qs[mrow * 40 + 8 * lg];
        ta.u[1] = *(const u32x4*)&Qs[mrow * 40 + 16 + 8 * lg];
#pragma unroll
        for (int nt = 0; nt < 4; ++nt) {
            int nrow = nt * 16 + lm;
            U16BF tb;
            tb.u[0] = *(const u32x4*)&Ks[nrow * 40 + 8 * lg];
            tb.u[1] = *(const u32x4*)&Ks[nrow * 40 + 16 + 8 * lg];
            v8f c = {};
            c = __builtin_amdgcn_wmma_f32_16x16x32_bf16(
                false, ta.v, false, tb.v, (short)0, c, false, false);
#pragma unroll
            for (int r = 0; r < 8; ++r)
                Sf[(w * 16 + r + 8 * lg) * 66 + nt * 16 + lm] = c[r];
        }
    }
    // ---- transpose V through LDS: Vt[d][key] (overlaps S stores; own data) ----
    {
        int d  = tid >> 2;               // 0..31
        int nq = (tid & 3) * 16;         // key base
#pragma unroll
        for (int i = 0; i < 16; ++i) {
            int n = nq + i;
            Vt[d * 72 + n] = Vr[n * 40 + d];
        }
    }
    __syncthreads();

    // ---- softmax per row (one thread per row; pad rows/cols -> 0) ----
    if (tid < 64) {
        int row = tid;
        if (row < N) {
            float mx = -1e30f;
            for (int n = 0; n < N; ++n) mx = fmaxf(mx, Sf[row * 66 + n]);
            float sum = 0.0f;
            for (int n = 0; n < N; ++n) sum += __expf(Sf[row * 66 + n] - mx);
            float inv = 1.0f / sum;
            for (int n = 0; n < 64; ++n) {
                float p = (n < N) ? __expf(Sf[row * 66 + n] - mx) * inv : 0.0f;
                Ps[row * 72 + n] = (__bf16)p;
            }
        } else {
            for (int n = 0; n < 64; ++n) Ps[row * 72 + n] = (__bf16)0.0f;
        }
    }
    __syncthreads();

    // ---- O = P V (K=64 in two x32 steps) + a*V, store ----
    {
        float av = aptr[0];
        int  mrow = w * 16 + lm;
        v8f  oacc[2];
        oacc[0] = {}; oacc[1] = {};
#pragma unroll
        for (int kk = 0; kk < 2; ++kk) {
            U16BF ta;
            ta.u[0] = *(const u32x4*)&Ps[mrow * 72 + kk * 32 + 8 * lg];
            ta.u[1] = *(const u32x4*)&Ps[mrow * 72 + kk * 32 + 16 + 8 * lg];
#pragma unroll
            for (int nt = 0; nt < 2; ++nt) {
                int drow = nt * 16 + lm;
                U16BF tb;
                tb.u[0] = *(const u32x4*)&Vt[drow * 72 + kk * 32 + 8 * lg];
                tb.u[1] = *(const u32x4*)&Vt[drow * 72 + kk * 32 + 16 + 8 * lg];
                oacc[nt] = __builtin_amdgcn_wmma_f32_16x16x32_bf16(
                    false, ta.v, false, tb.v, (short)0, oacc[nt], false, false);
            }
        }
#pragma unroll
        for (int nt = 0; nt < 2; ++nt) {
#pragma unroll
            for (int r = 0; r < 8; ++r) {
                int row = w * 16 + r + 8 * lg;
                if (row < N) {
                    int d = nt * 16 + lm;
                    float vv = (float)Vr[row * 40 + d];
                    ob[(tokbase + row) * 512 + ch + d] = (__bf16)(oacc[nt][r] + av * vv);
                }
            }
        }
    }
}

// =====================================================================
// Launch: 4 projection GEMMs -> attention -> output projection GEMM
// =====================================================================
extern "C" void kernel_launch(void* const* d_in, const int* in_sizes, int n_in,
                              void* d_out, int out_size, void* d_ws, size_t ws_size,
                              hipStream_t stream) {
    const float* x     = (const float*)d_in[0];
    const float* x2    = (const float*)d_in[1];
    const float* x3    = (const float*)d_in[2];
    const float* Wqkv  = (const float*)d_in[3];
    const float* bqkv  = (const float*)d_in[4];
    const float* Wqkv2 = (const float*)d_in[5];
    const float* bqkv2 = (const float*)d_in[6];
    const float* Wgw   = (const float*)d_in[7];
    const float* bgw   = (const float*)d_in[8];
    const float* Wproj = (const float*)d_in[9];
    const float* bproj = (const float*)d_in[10];
    const float* a     = (const float*)d_in[11];
    float* out = (float*)d_out;

    const int    M  = 1024 * 49;               // 50176 tokens (= 392 * 128)
    const size_t BE = (size_t)M * 512;         // elements per intermediate
    __bf16* kb = (__bf16*)d_ws;
    __bf16* qb = kb + BE;
    __bf16* vb = qb + BE;
    __bf16* gb = vb + BE;
    __bf16* ob = gb + BE;

    dim3 ggrid(4, 392), gblk(256);
    // k   = x  @ Wqkv[:,512:1024] + bqkv[512:1024]
    gemm_k512<float, __bf16><<<ggrid, gblk, 0, stream>>>(x,  Wqkv  + 512,  1536, bqkv  + 512,  kb);
    // q1  = x3 @ Wqkv2[:,0:512]   + bqkv2[0:512]
    gemm_k512<float, __bf16><<<ggrid, gblk, 0, stream>>>(x3, Wqkv2,        1536, bqkv2,        qb);
    // v1  = x3 @ Wqkv2[:,1024:]   + bqkv2[1024:]
    gemm_k512<float, __bf16><<<ggrid, gblk, 0, stream>>>(x3, Wqkv2 + 1024, 1536, bqkv2 + 1024, vb);
    // gw  = x2 @ Wgw + bgw
    gemm_k512<float, __bf16><<<ggrid, gblk, 0, stream>>>(x2, Wgw,          512,  bgw,          gb);
    // attention per (b,h)
    attn49<<<1024 * 16, 128, 0, stream>>>(qb, gb, kb, vb, a, ob);
    // out = attn_out @ Wproj + bproj  (f32 output)
    gemm_k512<__bf16, float><<<ggrid, gblk, 0, stream>>>(ob, Wproj,        512,  bproj,        out);
}
